// RNN_Variational_Decoder_71717363908646
// MI455X (gfx1250) — compile-verified, matrix operand order checked by reading, not run on hardware
//
#include <hip/hip_runtime.h>
#include <hip/hip_bf16.h>

// ---------------------------------------------------------------------------
// CDNA5 (gfx1250) fused LSTM decoder.
//  - 16 groups (batch tiles of 16 rows) x 16 WGs per group = 256 WGs, 256 thr.
//  - Each WG owns a 1/16 feature slice of all weights; the big gate-GEMM slice
//    (128x640 bf16, 160KB) is LDS-resident for the whole 512-step scan.
//  - All matmuls via v_wmma_f32_16x16x32_bf16 (f32 accumulate).
//  - Bulk global->LDS staging via the Tensor Data Mover (tensor_load_to_lds,
//    TENSORcnt) instead of VGPR round-trips; idle waves prefetch streamed
//    weight slices (global_prefetch_b8).
//  - Per-step cross-WG exchange (h / tanh-hidden / x=p1) through L2 with
//    agent-scope acquire/release group barriers (sense-reversing, in ws).
//  - Packed outputs written in-place using prefix[t]+b (lengths sorted desc).
// ---------------------------------------------------------------------------

typedef __attribute__((ext_vector_type(16))) __bf16 v16bf;
typedef __attribute__((ext_vector_type(8)))  float  v8f;
typedef __attribute__((ext_vector_type(4)))  unsigned u32x4;
typedef __attribute__((ext_vector_type(8)))  int      i32x8;
typedef __attribute__((ext_vector_type(4)))  int      i32x4;

#define K_B  256
#define K_T  512
#define K_H  512
#define K_O  128
#define K_MH 512
#define K_F  256
#define NGRP 16          // groups (batch tiles)
#define GWG  16          // workgroups per group
#define JS   32          // h-slice per WG

// ---- workspace layout (bytes) ----
#define OFF_PREFIX 0u          // int[513]
#define OFF_BARS   4096u       // uint[NGRP*2]
#define OFF_OFFP   8192u       // float[NGRP*16*16]
#define OFF_XH     24576u      // bf16[NGRP*10240]  (A-format: x K0..127 | h K128..639)
#define OFF_HID    352256u     // bf16[NGRP*2*8192] (p1h, p2h A-format)
#define OFF_CBUF   876544u     // float[K_B*K_H]
#define OFF_WG     1400832u    // bf16[16*81920]   gate-GEMM slices, WMMA B-tiles
#define OFF_W1     4022272u    // bf16[3*16*16384] p1/p2/off W1 slices, B-tiles
#define OFF_W2     5595136u    // bf16[2*8*8192]   p1/p2 W2 tiles, B-tiles

// element (m,k) -> half index inside A-format buffer (ktiles of 32)
__device__ __host__ inline int a_idx(int m, int k) {
  int kt = k >> 5, ko = k & 31;
  int lane = m + ((ko & 8) ? 16 : 0);
  int q = (ko & 7) + ((ko & 16) ? 8 : 0);
  return kt * 512 + lane * 16 + q;
}
// (lane,q) -> K-offset inside a 32-K tile (same striping for A and B tiles)
__device__ inline int bk0(int lane, int q) {
  return q + ((q & 8) ? 8 : 0) + ((lane & 16) ? 8 : 0);
}

__device__ inline v8f wmma_bf16(v16bf a, v16bf b, v8f c) {
  return __builtin_amdgcn_wmma_f32_16x16x32_bf16(false, a, false, b, (short)0, c,
                                                 false, false);
}
__device__ inline float sigm(float x) { return 1.f / (1.f + __expf(-x)); }

// ---- Tensor Data Mover: 2D tile (ndw_x dwords x ndw_y rows), contiguous ----
// D# per CDNA5 ISA: group0 = {count=1, lds_addr,
// global_addr[56:0], type=2}; group1 = {data_size=4B, tensor/tile dims,
// dim0 stride}. Groups 2/3 zero (<=2D tensor). Issued by one wave; EXEC
// ignored by TDM; completion via s_wait_tensorcnt. 6-arg builtin form
// (clang-23 / therock-10.0 headers).
__device__ inline void tdm_load_2d(unsigned lds_off, const void* gsrc,
                                   int ndw_x, int ndw_y) {
  unsigned long long ga = (unsigned long long)(size_t)gsrc;
  u32x4 g0;
  g0[0] = 1u;                                        // count = 1 (valid D#)
  g0[1] = lds_off;                                   // lds_addr (bytes)
  g0[2] = (unsigned)ga;                              // global_addr[31:0]
  g0[3] = (unsigned)((ga >> 32) & 0x01FFFFFFu) | 0x80000000u;  // [56:32]|type=2
  i32x8 g1;
  g1[0] = 2 << 16;                                   // data_size = 4 bytes
  g1[1] = (ndw_x & 0xFFFF) << 16;                    // tensor_dim0[15:0]
  g1[2] = ((unsigned)ndw_x >> 16) | ((unsigned)(ndw_y & 0xFFFF) << 16);
  g1[3] = (ndw_x << 16);                             // tile_dim0 (dim1 hi = 0)
  g1[4] = ndw_y & 0xFFFF;                            // tile_dim1 (tile_dim2=0)
  g1[5] = ndw_x;                                     // tensor_dim0_stride lo
  g1[6] = 0;
  g1[7] = 0;
  i32x4 z4 = {0, 0, 0, 0};
  i32x8 z8 = {0, 0, 0, 0, 0, 0, 0, 0};
  __builtin_amdgcn_tensor_load_to_lds(g0, g1, z4, z4, z8, 0);
}

// ---------------------------------------------------------------------------
__global__ void setup_prefix(const int* __restrict__ lens, int* __restrict__ prefix,
                             unsigned* __restrict__ bars) {
  __shared__ int cnt[K_T];
  int t = threadIdx.x;
  int c = 0;
  for (int b = 0; b < K_B; ++b) c += (lens[b] > t) ? 1 : 0;
  cnt[t] = c;
  if (t < NGRP * 2) bars[t] = 0u;
  __syncthreads();
  if (t == 0) {
    int s = 0;
    for (int i = 0; i < K_T; ++i) { prefix[i] = s; s += cnt[i]; }
    prefix[K_T] = s;
  }
}

__global__ void zero_xh(uint4* __restrict__ p, int n) {
  int i = blockIdx.x * blockDim.x + threadIdx.x;
  if (i < n) p[i] = uint4{0, 0, 0, 0};
}

// ---------------------------------------------------------------------------
__global__ void pack_weights(const float* __restrict__ W_ih, const float* __restrict__ W_hh,
                             const float* __restrict__ p1W1, const float* __restrict__ p2W1,
                             const float* __restrict__ offW1,
                             const float* __restrict__ p1W2, const float* __restrict__ p2W2,
                             __bf16* __restrict__ wg, __bf16* __restrict__ w1,
                             __bf16* __restrict__ w2) {
  const long NWG = 16L * 81920;   // gate slices
  const long NW1 = 3L * 16 * 16384;
  const long NW2 = 2L * 8 * 8192;
  for (long i = blockIdx.x * 256L + threadIdx.x; i < NWG + NW1 + NW2;
       i += (long)gridDim.x * 256L) {
    if (i < NWG) {
      int wslot = (int)(i / 81920); int r = (int)(i % 81920);
      int tile = r >> 9, within = r & 511, lane = within >> 4, q = within & 15;
      int kt = tile >> 3, nt = tile & 7;
      int ncol = nt * 16 + (lane & 15);              // 0..127 = [i f g o] x 32
      int k = kt * 32 + bk0(lane, q);                // 0..639
      int grow = (ncol >> 5) * 512 + wslot * 32 + (ncol & 31);
      float v = (k < 128) ? W_ih[grow * 128 + k] : W_hh[grow * 512 + (k - 128)];
      wg[i] = (__bf16)v;
    } else if (i < NWG + NW1) {
      long e = i - NWG;
      int mat = (int)(e / (16 * 16384)); int r = (int)(e % (16 * 16384));
      int wslot = r / 16384; int r2 = r % 16384;
      int tile = r2 >> 9, within = r2 & 511, lane = within >> 4, q = within & 15;
      int kt = tile >> 1, ntl = tile & 1;
      int row = wslot * 32 + ntl * 16 + (lane & 15);
      int k = kt * 32 + bk0(lane, q);
      const float* src = (mat == 0) ? p1W1 : ((mat == 1) ? p2W1 : offW1);
      w1[e] = (__bf16)src[row * 512 + k];
    } else {
      long e = i - NWG - NW1;
      int mat = (int)(e / 65536); int r = (int)(e % 65536);
      int nt = r / 8192; int r2 = r % 8192;
      int within = r2 & 511, lane = within >> 4, q = within & 15;
      int kt = r2 >> 9;
      int row = nt * 16 + (lane & 15);
      int k = kt * 32 + bk0(lane, q);
      const float* src = (mat == 0) ? p1W2 : p2W2;
      w2[e] = (__bf16)src[row * 512 + k];
    }
  }
}

// hid = features @ W_f2h.T + b ; even cols -> h0 (bf16, A-format), odd -> c0
__global__ void init_state(const float* __restrict__ feat, const float* __restrict__ Wf2h,
                           const float* __restrict__ bf2h, __bf16* __restrict__ xh,
                           float* __restrict__ cbuf) {
  int idx = blockIdx.x * blockDim.x + threadIdx.x;  // K_B*1024
  int b = idx >> 10, col = idx & 1023;
  float acc = bf2h[col];
  const float* fr = feat + b * K_F;
  const float* wr = Wf2h + col * K_F;
  for (int k = 0; k < K_F; ++k) acc += fr[k] * wr[k];
  int j = col >> 1, grp = b >> 4, m = b & 15;
  if ((col & 1) == 0) xh[grp * 10240 + a_idx(m, 128 + j)] = (__bf16)acc;
  else cbuf[b * K_H + j] = acc;
}

// ---------------------------------------------------------------------------
__device__ inline void group_barrier(unsigned* cnt, unsigned* gen) {
  __syncthreads();
  if (threadIdx.x == 0) {
    unsigned g = __hip_atomic_load(gen, __ATOMIC_RELAXED, __HIP_MEMORY_SCOPE_AGENT);
    unsigned o = __hip_atomic_fetch_add(cnt, 1u, __ATOMIC_ACQ_REL, __HIP_MEMORY_SCOPE_AGENT);
    if (o == GWG - 1) {
      __hip_atomic_store(cnt, 0u, __ATOMIC_RELAXED, __HIP_MEMORY_SCOPE_AGENT);
      __hip_atomic_fetch_add(gen, 1u, __ATOMIC_RELEASE, __HIP_MEMORY_SCOPE_AGENT);
    } else {
      while (__hip_atomic_load(gen, __ATOMIC_ACQUIRE, __HIP_MEMORY_SCOPE_AGENT) == g)
        __builtin_amdgcn_s_sleep(2);
    }
  }
  __syncthreads();
}

__global__ __launch_bounds__(256, 1) void rnn_main(
    const float* __restrict__ b_ih, const float* __restrict__ b_hh,
    const float* __restrict__ p1b1, const float* __restrict__ p2b1,
    const float* __restrict__ offb1, const float* __restrict__ offW2,
    const float* __restrict__ p1b2, const float* __restrict__ p2b2,
    const float* __restrict__ offb2, const int* __restrict__ lens,
    const __bf16* __restrict__ wgp, const __bf16* __restrict__ w1p,
    const __bf16* __restrict__ w2p, __bf16* __restrict__ g_xh,
    __bf16* __restrict__ g_hid, const float* __restrict__ cbuf,
    const int* __restrict__ prefix_g, float* __restrict__ offp,
    unsigned* __restrict__ bars, float* __restrict__ out_base) {
  const int group = blockIdx.x >> 4;
  const int ws_ = blockIdx.x & 15;  // workgroup slot within group
  const int tid = threadIdx.x;
  const int lane = tid & 31;
  const int wv = tid >> 5;  // wave 0..7

  __shared__ __align__(32) __bf16 sWg[20 * 8 * 512];  // 160KB resident gate slice
  __shared__ __align__(32) __bf16 sXH[20 * 512];      // 20KB  x|h A-format
  __shared__ __align__(32) unsigned char sUnion[16384]; // gates f32 (A/B) | hid bf16 (D)
  __shared__ float sC[512];
  __shared__ float sProd[512];
  __shared__ float sGBias[128];
  __shared__ float sB1[96];
  __shared__ float sOffW2[32];
  __shared__ float sB2[16];
  __shared__ int sLen[16];

  float* sGates = (float*)sUnion;
  __bf16* sHid = (__bf16*)sUnion;

  __bf16* gxh = g_xh + group * 10240;
  __bf16* ghid1 = g_hid + group * 16384;
  __bf16* ghid2 = ghid1 + 8192;
  unsigned* cntp = bars + group * 2;
  unsigned* genp = cntp + 1;

  const unsigned ldsWg = (unsigned)(size_t)(void*)sWg;
  const unsigned ldsXH = (unsigned)(size_t)(void*)sXH;
  const unsigned ldsHid = (unsigned)(size_t)(void*)sUnion;

  // ---- one-time LDS init: big copies via Tensor Data Mover ----
  if (wv == 0) {  // 160KB gate-GEMM slice: 4096 dwords x 10 rows
    tdm_load_2d(ldsWg, wgp + (size_t)ws_ * 81920, 4096, 10);
    __builtin_amdgcn_s_wait_tensorcnt(0);
  } else if (wv == 1) {  // 20KB x|h state image
    tdm_load_2d(ldsXH, gxh, 5120, 1);
    __builtin_amdgcn_s_wait_tensorcnt(0);
  }
  for (int e = tid; e < 512; e += 256) {
    int m = e >> 5, jj = e & 31;
    sC[e] = cbuf[(group * 16 + m) * K_H + ws_ * 32 + jj];
  }
  if (tid < 128) {
    int grow = (tid >> 5) * 512 + ws_ * 32 + (tid & 31);
    sGBias[tid] = b_ih[grow] + b_hh[grow];
  }
  if (tid < 32) {
    sB1[tid] = p1b1[ws_ * 32 + tid];
    sB1[32 + tid] = p2b1[ws_ * 32 + tid];
    sB1[64 + tid] = offb1[ws_ * 32 + tid];
    sOffW2[tid] = offW2[ws_ * 32 + tid];
  }
  if (tid < 16) {
    sLen[tid] = lens[group * 16 + tid];
    int ntD = ws_ & 7;
    sB2[tid] = (ws_ < 8) ? p1b2[ntD * 16 + tid] : p2b2[ntD * 16 + tid];
  }
  __syncthreads();

  const int lenmax = sLen[0];
  const int NP = prefix_g[K_T];
  const float offb2v = offb2[0];
  float* out_p1 = out_base;
  float* out_p2 = out_base + (size_t)NP * 128;
  float* out_off = out_base + (size_t)NP * 256;
  float* out_cp = out_base + (size_t)NP * 257;

  for (int t = 0; t < K_T; ++t) {
    if (t >= lenmax) break;
    const int pt = prefix_g[t];

    // ---- Phase A: gate slice = [x|h] @ Wg_slice.T (8 N-tiles, 1 per wave) ----
    {
      v8f acc = {0.f, 0.f, 0.f, 0.f, 0.f, 0.f, 0.f, 0.f};
      const int nt = wv;
      for (int kt = 0; kt < 20; ++kt) {
        v16bf a = *(const v16bf*)(sXH + kt * 512 + lane * 16);
        v16bf b = *(const v16bf*)(sWg + (kt * 8 + nt) * 512 + lane * 16);
        acc = wmma_bf16(a, b, acc);
      }
      int col = nt * 16 + (lane & 15);
      float bv = sGBias[col];
      int mb = (lane >> 4) * 8;
#pragma unroll
      for (int r = 0; r < 8; ++r) sGates[(mb + r) * 128 + col] = acc[r] + bv;
    }
    __syncthreads();

    // ---- Phase B: LSTM cell on our 32 h-indices, publish h slice ----
    for (int e = tid; e < 512; e += 256) {
      int m = e >> 5, jj = e & 31;
      float gi = sGates[m * 128 + jj];
      float gf = sGates[m * 128 + 32 + jj];
      float gg = sGates[m * 128 + 64 + jj];
      float go = sGates[m * 128 + 96 + jj];
      float c_old = sC[e];
      float cn = sigm(gf) * c_old + sigm(gi) * tanhf(gg);
      float hn = sigm(go) * tanhf(cn);
      int j = ws_ * 32 + jj;
      if (t >= sLen[m]) { cn = c_old; hn = (float)sXH[a_idx(m, 128 + j)]; }
      sC[e] = cn;
      gxh[a_idx(m, 128 + j)] = (__bf16)hn;
    }
    group_barrier(cntp, genp);
    // gather full updated h (16KB) via TDM; other waves prefetch W1 stream
    if (wv == 0) {
      tdm_load_2d(ldsXH + 4096u, (const void*)(gxh + 2048), 4096, 1);
      __builtin_amdgcn_s_wait_tensorcnt(0);
    } else if (wv >= 2) {
      int mat = (wv - 2) >> 1;
      int i0 = ((wv - 2) & 1) * 32 + lane;  // 0..63 per mat
      const char* pf =
          (const char*)(w1p + (size_t)(mat * 16 + ws_) * 16384) + i0 * 512;
      __builtin_prefetch(pf, 0, 0);
    }
    __syncthreads();

    // ---- Phase C: p1/p2/off hidden slices (6 N-tiles on waves 0..5) ----
    if (wv < 6) {
      int mat = wv >> 1, ntl = wv & 1;
      v8f acc = {0.f, 0.f, 0.f, 0.f, 0.f, 0.f, 0.f, 0.f};
      const __bf16* wsrc = w1p + (size_t)(mat * 16 + ws_) * 16384;
      for (int kt = 0; kt < 16; ++kt) {
        v16bf a = *(const v16bf*)(sXH + (4 + kt) * 512 + lane * 16);
        v16bf b = *(const v16bf*)(wsrc + (kt * 2 + ntl) * 512 + lane * 16);
        acc = wmma_bf16(a, b, acc);
      }
      int cl = ntl * 16 + (lane & 15);
      float bv = sB1[mat * 32 + cl];
      int mb = (lane >> 4) * 8;
      if (mat < 2) {
        __bf16* gh = (mat == 0) ? ghid1 : ghid2;
        int j2 = ws_ * 32 + cl;
#pragma unroll
        for (int r = 0; r < 8; ++r) {
          float th = tanhf(acc[r] + bv);
          gh[a_idx(mb + r, j2)] = (__bf16)th;
        }
      } else {
        float w2v = sOffW2[cl];
#pragma unroll
        for (int r = 0; r < 8; ++r) {
          float th = tanhf(acc[r] + bv);
          sProd[(mb + r) * 32 + cl] = th * w2v;
        }
      }
    } else {  // idle waves: warm the W2 tile for phase D (16KB, 64 x 256B)
      const char* pf =
          (const char*)(w2p + (size_t)(((ws_ < 8) ? 0 : 8) + (ws_ & 7)) * 8192);
      int i0 = (wv - 6) * 32 + lane;
      __builtin_prefetch(pf + i0 * 256, 0, 0);
    }
    __syncthreads();
    if (tid < 16) {  // deterministic per-WG offsets partial -> global slot
      float s = 0.f;
      for (int c = 0; c < 32; ++c) s += sProd[tid * 32 + c];
      offp[group * 256 + ws_ * 16 + tid] = s;
    }
    group_barrier(cntp, genp);
    // pull the tanh-hidden image this WG needs for its W2 tile (16KB, TDM)
    if (wv == 0) {
      tdm_load_2d(ldsHid, (const void*)((ws_ < 8) ? ghid1 : ghid2), 4096, 1);
      __builtin_amdgcn_s_wait_tensorcnt(0);
    }
    __syncthreads();

    // ---- Phase D: one output N-tile per WG (p1 for wg<8 feeds next x) ----
    if (wv == 0) {
      int ntD = ws_ & 7;
      v8f acc = {0.f, 0.f, 0.f, 0.f, 0.f, 0.f, 0.f, 0.f};
      const __bf16* wsrc = w2p + (size_t)(((ws_ < 8) ? 0 : 8) + ntD) * 8192;
      for (int kt = 0; kt < 16; ++kt) {
        v16bf a = *(const v16bf*)(sHid + kt * 512 + lane * 16);
        v16bf b = *(const v16bf*)(wsrc + kt * 512 + lane * 16);
        acc = wmma_bf16(a, b, acc);
      }
      int cl = ntD * 16 + (lane & 15);
      float bv = sB2[lane & 15];
      int mb = (lane >> 4) * 8;
#pragma unroll
      for (int r = 0; r < 8; ++r) {
        int m = mb + r;
        float val = acc[r] + bv;
        if (ws_ < 8) gxh[a_idx(m, cl)] = (__bf16)val;  // x_{t+1}
        if (t < sLen[m]) {
          size_t pos = (size_t)(pt + group * 16 + m);
          if (ws_ < 8) {
            out_p1[pos * 128 + cl] = val;
            out_cp[pos * 128 + cl] = val;
          } else {
            out_p2[pos * 128 + cl] = val;
          }
        }
      }
    }
    if (ws_ == 0 && wv == 1 && lane < 16) {  // single-writer offsets reduce
      float s = 0.f;
      for (int sl = 0; sl < 16; ++sl) s += offp[group * 256 + sl * 16 + lane];
      if (t < sLen[lane]) out_off[pt + group * 16 + lane] = s + offb2v;
    }
    group_barrier(cntp, genp);
    {  // reload x part (2048 halfs = 4KB) — small, plain vector copy
      const uint4* src = (const uint4*)gxh;
      uint4* dst = (uint4*)sXH;
      if (tid < 256) dst[tid] = src[tid];
    }
    __syncthreads();
  }
}

// ---------------------------------------------------------------------------
extern "C" void kernel_launch(void* const* d_in, const int* in_sizes, int n_in,
                              void* d_out, int out_size, void* d_ws, size_t ws_size,
                              hipStream_t stream) {
  const float* features = (const float*)d_in[0];
  const float* W_f2h = (const float*)d_in[1];
  const float* b_f2h = (const float*)d_in[2];
  const float* W_ih = (const float*)d_in[3];
  const float* W_hh = (const float*)d_in[4];
  const float* b_ih = (const float*)d_in[5];
  const float* b_hh = (const float*)d_in[6];
  const float* p1W1 = (const float*)d_in[7];
  const float* p1b1 = (const float*)d_in[8];
  const float* p1W2 = (const float*)d_in[9];
  const float* p1b2 = (const float*)d_in[10];
  const float* p2W1 = (const float*)d_in[11];
  const float* p2b1 = (const float*)d_in[12];
  const float* p2W2 = (const float*)d_in[13];
  const float* p2b2 = (const float*)d_in[14];
  const float* offW1 = (const float*)d_in[15];
  const float* offb1 = (const float*)d_in[16];
  const float* offW2 = (const float*)d_in[17];
  const float* offb2 = (const float*)d_in[18];
  const int* lens = (const int*)d_in[19];
  (void)in_sizes; (void)n_in; (void)out_size; (void)ws_size;

  char* w = (char*)d_ws;
  int* prefix = (int*)(w + OFF_PREFIX);
  unsigned* bars = (unsigned*)(w + OFF_BARS);
  float* offp = (float*)(w + OFF_OFFP);
  __bf16* g_xh = (__bf16*)(w + OFF_XH);
  __bf16* g_hid = (__bf16*)(w + OFF_HID);
  float* cbuf = (float*)(w + OFF_CBUF);
  __bf16* wgp = (__bf16*)(w + OFF_WG);
  __bf16* w1p = (__bf16*)(w + OFF_W1);
  __bf16* w2p = (__bf16*)(w + OFF_W2);

  setup_prefix<<<1, 512, 0, stream>>>(lens, prefix, bars);
  zero_xh<<<80, 256, 0, stream>>>((uint4*)g_xh, 20480);
  pack_weights<<<4352, 256, 0, stream>>>(W_ih, W_hh, p1W1, p2W1, offW1, p1W2, p2W2,
                                         wgp, w1p, w2p);
  init_state<<<1024, 256, 0, stream>>>(features, W_f2h, b_f2h, g_xh, cbuf);
  rnn_main<<<NGRP * GWG, 256, 0, stream>>>(b_ih, b_hh, p1b1, p2b1, offb1, offW2, p1b2,
                                           p2b2, offb2, lens, wgp, w1p, w2p, g_xh,
                                           g_hid, cbuf, prefix, offp, bars,
                                           (float*)d_out);
}
